// DAGN_76656576299456
// MI455X (gfx1250) — matrix operand order repaired and checked by reading, not compiled, not run on hardware
//
#include <hip/hip_runtime.h>
#include <hip/hip_bf16.h>
#include <math.h>

// ---------------------------------------------------------------------------
// DAGN forward for MI455X (gfx1250, wave32, WMMA).
//  - all input projections hoisted into big bf16 WMMA GEMMs
//  - GEMM tiles staged with GLOBAL_LOAD_ASYNC_TO_LDS_B128 (ASYNCcnt) and
//    double-buffered LDS: one barrier per K-step, no VGPR staging round-trip
//  - persistent LSTM/GRU scan kernels keep state in LDS, do h@Whh with
//    v_wmma_f32_16x16x32_bf16 (M=16 = batch -> one WMMA tile row)
// ---------------------------------------------------------------------------

typedef unsigned short u16;
typedef unsigned int   u32;
typedef __attribute__((ext_vector_type(16))) __bf16 v16bf;
typedef __attribute__((ext_vector_type(8)))  float  v8f;

#define B_   16
#define S_   512
#define E_   512
#define DH   512
#define H2_  256
#define NSP  257      // span-node bound (S/2+1)
#define NPAD 272      // padded to a multiple of 16 so B*NPAD = 4352 = 68*64

__device__ __forceinline__ u16 f2bf(float f) {
  u32 u = __float_as_uint(f);
  u32 r = (u + 0x7FFFu + ((u >> 16) & 1u)) >> 16;   // RNE
  return (u16)r;
}
__device__ __forceinline__ float sigf(float x) { return 1.f / (1.f + __expf(-x)); }

union FragBF { v16bf v; uint4 q[2]; };

// gfx1250 async global->LDS copy (16B per lane), tracked by ASYNCcnt.
__device__ __forceinline__ void async_ld_b128(u32 lds_off, const void* gaddr) {
  asm volatile("global_load_async_to_lds_b128 %0, %1, off"
               :: "v"(lds_off), "v"(gaddr) : "memory");
}
__device__ __forceinline__ void wait_async0() {
  asm volatile("s_wait_asynccnt 0" ::: "memory");
}
__device__ __forceinline__ u32 lds_addr(const void* p) {
  return (u32)(unsigned long long)p;   // generic addr low 32 bits = LDS offset
}

// ---------------------------------------------------------------------------
// Generic bf16 GEMM: C[M,N] = A[M,K] * B[K,N] (+bias[N]); optional bf16 copy.
// A,B row-major bf16; M,N multiples of 64; K multiple of 32.
// Block = 256 threads = 8 waves; tile 64x64x32; each wave owns 16x32 of C.
// Double-buffered LDS, async tile staging. Fragment layouts per ISA 7.12.2.
// ---------------------------------------------------------------------------
#define GBM 64
#define GBN 64
#define GBK 32

__global__ __launch_bounds__(256) void gemm_bf16_wmma(
    const u16* __restrict__ A, const u16* __restrict__ Bm,
    const float* __restrict__ bias, float* __restrict__ C,
    u16* __restrict__ Cbf, int M, int Nn, int K)
{
  __shared__ __align__(16) u16 sA[2][GBM][GBK];   // 2 x 4 KB
  __shared__ __align__(16) u16 sB[2][GBK][GBN];   // 2 x 4 KB
  const int tid  = threadIdx.x;
  const int lane = tid & 31;
  const int wave = tid >> 5;
  const int m0 = blockIdx.x * GBM;
  const int n0 = blockIdx.y * GBN;
  const int wm = (wave & 3) * 16;   // M sub-tile
  const int wn = (wave >> 2) * 32;  // N sub-tile (2 accumulators)

  v8f acc0 = {}, acc1 = {};
  if (bias) {
    float b0v = bias[n0 + wn + (lane & 15)];
    float b1v = bias[n0 + wn + 16 + (lane & 15)];
    for (int r = 0; r < 8; ++r) { acc0[r] = b0v; acc1[r] = b1v; }
  }
  const int arow = tid >> 2, apart = tid & 3;  // 64 rows x 4 x 8 elems
  const int brow = tid >> 3, bpart = tid & 7;  // 32 rows x 8 x 8 elems

  auto stage = [&](int buf, int k0) {
    async_ld_b128(lds_addr(&sA[buf][arow][apart * 8]),
                  &A[(size_t)(m0 + arow) * K + k0 + apart * 8]);
    async_ld_b128(lds_addr(&sB[buf][brow][bpart * 8]),
                  &Bm[(size_t)(k0 + brow) * Nn + n0 + bpart * 8]);
  };

  const int KT = K / GBK;
  stage(0, 0);
  wait_async0();
  __syncthreads();                       // buffer 0 ready for all waves

  int buf = 0;
  for (int kt = 0; kt < KT; ++kt) {
    const int more = (kt + 1 < KT);
    if (more) stage(buf ^ 1, (kt + 1) * GBK);   // overlap copy with compute
    FragBF fa, fb0, fb1;
    const int am = wm + (lane & 15);
    const int ak = (lane >> 4) * 8;
    fa.q[0] = *(const uint4*)&sA[buf][am][ak];       // K[base .. base+8)
    fa.q[1] = *(const uint4*)&sA[buf][am][ak + 16];  // K[base+16 .. base+24)
    for (int s = 0; s < 16; ++s) {                   // B: lane%16 = N col, elems over K (table order)
      const int kr = ak + (s < 8 ? s : s + 8);
      ((u16*)&fb0)[s] = sB[buf][kr][wn + (lane & 15)];
      ((u16*)&fb1)[s] = sB[buf][kr][wn + 16 + (lane & 15)];
    }
    acc0 = __builtin_amdgcn_wmma_f32_16x16x32_bf16(false, fa.v, false, fb0.v,
                                                   (short)0, acc0, false, false);
    acc1 = __builtin_amdgcn_wmma_f32_16x16x32_bf16(false, fa.v, false, fb1.v,
                                                   (short)0, acc1, false, false);
    if (more) {
      wait_async0();                     // own async ops for next buffer done
      __syncthreads();                   // all waves staged + done reading buf
    }
    buf ^= 1;
  }
  const int rbase = m0 + wm + 8 * (lane >> 4);
  const int ncol  = n0 + wn + (lane & 15);
  for (int r = 0; r < 8; ++r) {
    size_t i0 = (size_t)(rbase + r) * Nn + ncol;
    float v0 = acc0[r]; C[i0] = v0; if (Cbf) Cbf[i0] = f2bf(v0);
    size_t i1 = i0 + 16;
    float v1 = acc1[r]; C[i1] = v1; if (Cbf) Cbf[i1] = f2bf(v1);
  }
}

// ---------------------------------------------------------------------------
// Persistent BiLSTM scan. grid.x = direction. 1024 threads = 32 waves.
// gi = precomputed x@Wih + bih + bhh, [B][S][1024].
// Per step: gh = h(16x256,bf16 in LDS) @ WhhT(256x1024,bf16 L2-resident),
// one WMMA N-tile pair per wave, gates through LDS. Next-step gi prefetched.
// ---------------------------------------------------------------------------
__global__ __launch_bounds__(1024) void lstm_scan(
    const float* __restrict__ gi_f, const float* __restrict__ gi_b,
    const u16* __restrict__ WT_f, const u16* __restrict__ WT_b,
    float* __restrict__ out)
{
  extern __shared__ char smem[];
  u16*   hbf  = (u16*)smem;                  // [16][256]  8 KB  (bf16 h)
  float* cst  = (float*)(smem + 8 * 1024);   // [16][256] 16 KB  (c state)
  float* gbuf = (float*)(smem + 24 * 1024);  // [16][1024] 64 KB (gates)
  const int dir  = blockIdx.x;
  const float* gi = dir ? gi_b : gi_f;
  const u16* WhhT = dir ? WT_b : WT_f;
  const int tid = threadIdx.x, lane = tid & 31, wave = tid >> 5;

  for (int i = tid; i < 16 * 256; i += 1024) { hbf[i] = 0; cst[i] = 0.f; }
  __syncthreads();

  for (int step = 0; step < S_; ++step) {
    const int t = dir ? (S_ - 1 - step) : step;
    // phase 1: gh = h @ WhhT, 64 N-tiles over 32 waves
    for (int nt = 0; nt < 2; ++nt) {
      const int nbase = (wave * 2 + nt) * 16;
      v8f acc = {};
      const int am = lane & 15;
      const int ak = (lane >> 4) * 8;
      for (int kk = 0; kk < 8; ++kk) {
        const int kb = kk * 32;
        FragBF fa, fb;
        fa.q[0] = *(const uint4*)&hbf[am * 256 + kb + ak];
        fa.q[1] = *(const uint4*)&hbf[am * 256 + kb + ak + 16];
        for (int s = 0; s < 16; ++s) {
          const int kr = kb + ak + (s < 8 ? s : s + 8);
          ((u16*)&fb)[s] = WhhT[(size_t)kr * 1024 + nbase + am];
        }
        acc = __builtin_amdgcn_wmma_f32_16x16x32_bf16(false, fa.v, false, fb.v,
                                                      (short)0, acc, false, false);
      }
      const int mb = 8 * (lane >> 4);
      const int nc = nbase + (lane & 15);
      for (int r = 0; r < 8; ++r) {
        const int m = mb + r;
        gbuf[m * 1024 + nc] = acc[r] + gi[((size_t)m * S_ + t) * 1024 + nc];
      }
    }
    __syncthreads();
    // phase 2: gate nonlinearity + state update (4096 cells / 1024 threads)
    for (int rep = 0; rep < 4; ++rep) {
      const int idx = tid + rep * 1024;
      const int b = idx >> 8, j = idx & 255;
      if (rep == 0 && step + 1 < S_) {   // prefetch next step's gate inputs
        const int tn = dir ? (S_ - 2 - step) : (step + 1);
        __builtin_prefetch(&gi[((size_t)b * S_ + tn) * 1024 + j * 4], 0, 1);
      }
      const float ig = gbuf[b * 1024 + j];
      const float fg = gbuf[b * 1024 + 256 + j];
      const float gg = gbuf[b * 1024 + 512 + j];
      const float og = gbuf[b * 1024 + 768 + j];
      float c = sigf(fg) * cst[b * 256 + j] + sigf(ig) * tanhf(gg);
      float h = sigf(og) * tanhf(c);
      cst[b * 256 + j] = c;
      hbf[b * 256 + j] = f2bf(h);
      out[((size_t)b * S_ + t) * DH + dir * H2_ + j] = h;
    }
    __syncthreads();
  }
}

// ---------------------------------------------------------------------------
// Persistent BiGRU scan. grid.x = direction. gi = x@Wih + bih, [B][S][768].
// gh (WMMA) kept separate from gi so the reset gate can scale hn only.
// ---------------------------------------------------------------------------
__global__ __launch_bounds__(1024) void gru_scan(
    const float* __restrict__ gi_f, const float* __restrict__ gi_b,
    const u16* __restrict__ WT_f, const u16* __restrict__ WT_b,
    const float* __restrict__ bhh_f, const float* __restrict__ bhh_b,
    float* __restrict__ out)
{
  extern __shared__ char smem[];
  u16*   hbf  = (u16*)smem;                  // [16][256]  8 KB
  float* hst  = (float*)(smem + 8 * 1024);   // [16][256] 16 KB (f32 h)
  float* gbuf = (float*)(smem + 24 * 1024);  // [16][768] 48 KB (gh + bhh)
  const int dir = blockIdx.x;
  const float* gi = dir ? gi_b : gi_f;
  const u16* WhhT = dir ? WT_b : WT_f;
  const float* bhh = dir ? bhh_b : bhh_f;
  const int tid = threadIdx.x, lane = tid & 31, wave = tid >> 5;

  for (int i = tid; i < 16 * 256; i += 1024) { hbf[i] = 0; hst[i] = 0.f; }
  __syncthreads();

  for (int step = 0; step < S_; ++step) {
    const int t = dir ? (S_ - 1 - step) : step;
    for (int ntile = wave; ntile < 48; ntile += 32) {
      const int nbase = ntile * 16;
      v8f acc = {};
      const int am = lane & 15;
      const int ak = (lane >> 4) * 8;
      for (int kk = 0; kk < 8; ++kk) {
        const int kb = kk * 32;
        FragBF fa, fb;
        fa.q[0] = *(const uint4*)&hbf[am * 256 + kb + ak];
        fa.q[1] = *(const uint4*)&hbf[am * 256 + kb + ak + 16];
        for (int s = 0; s < 16; ++s) {
          const int kr = kb + ak + (s < 8 ? s : s + 8);
          ((u16*)&fb)[s] = WhhT[(size_t)kr * 768 + nbase + am];
        }
        acc = __builtin_amdgcn_wmma_f32_16x16x32_bf16(false, fa.v, false, fb.v,
                                                      (short)0, acc, false, false);
      }
      const int mb = 8 * (lane >> 4);
      const int nc = nbase + (lane & 15);
      for (int r = 0; r < 8; ++r) {
        const int m = mb + r;
        gbuf[m * 768 + nc] = acc[r] + bhh[nc];
      }
    }
    __syncthreads();
    for (int rep = 0; rep < 4; ++rep) {
      const int idx = tid + rep * 1024;
      const int b = idx >> 8, j = idx & 255;
      if (rep == 0 && step + 1 < S_) {   // prefetch next step's gate inputs
        const int tn = dir ? (S_ - 2 - step) : (step + 1);
        __builtin_prefetch(&gi[((size_t)b * S_ + tn) * 768 + j * 3], 0, 1);
      }
      const size_t gbase = ((size_t)b * S_ + t) * 768;
      float r = sigf(gi[gbase + j]       + gbuf[b * 768 + j]);
      float z = sigf(gi[gbase + 256 + j] + gbuf[b * 768 + 256 + j]);
      float n = tanhf(gi[gbase + 512 + j] + r * gbuf[b * 768 + 512 + j]);
      float hp = hst[b * 256 + j];
      float h = (1.f - z) * n + z * hp;
      hst[b * 256 + j] = h;
      hbf[b * 256 + j] = f2bf(h);
      out[((size_t)b * S_ + t) * DH + dir * H2_ + j] = h;
    }
    __syncthreads();
  }
}

// --------------------------- small utility kernels -------------------------

__global__ void transp_bf16(const float* __restrict__ w, u16* __restrict__ wt,
                            int Nn, int K) {  // w [Nn][K] f32 -> wt [K][Nn] bf16
  size_t i = (size_t)blockIdx.x * blockDim.x + threadIdx.x;
  if (i < (size_t)Nn * K) {
    int n = (int)(i / K), k = (int)(i % K);
    wt[(size_t)k * Nn + n] = f2bf(w[i]);
  }
}

__global__ void conv_bf16(const float* __restrict__ x, u16* __restrict__ y, size_t n) {
  size_t i = (size_t)blockIdx.x * blockDim.x + threadIdx.x;
  if (i < n) y[i] = f2bf(x[i]);
}

__global__ void add_vec(const float* a, const float* b, float* o, int n) {
  int i = blockIdx.x * blockDim.x + threadIdx.x;
  if (i < n) o[i] = a[i] + b[i];
}

__global__ void embed_bf(const int* __restrict__ ids, const float* __restrict__ emb,
                         u16* __restrict__ out) {
  int bs = blockIdx.x;
  int id = ids[bs];
  const float* src = &emb[(size_t)id * E_];
  u16* dst = &out[(size_t)bs * E_];
  for (int d = threadIdx.x; d < E_; d += blockDim.x) dst[d] = f2bf(src[d]);
}

// serial per-batch span/edge construction (tiny: 16 x 512)
__global__ void span_build(const int* __restrict__ am, const int* __restrict__ spl,
                           int* __restrict__ sid, int* __restrict__ member,
                           int* __restrict__ nnodes, float* __restrict__ ga,
                           float* __restrict__ gp, float* __restrict__ deg) {
  const int b = blockIdx.x;
  if (threadIdx.x) return;
  int span = -1, nspans = 0, pnm = 0;
  for (int s = 0; s < S_; ++s) {
    int valid = am[b * S_ + s] > 0;
    int mark = (spl[b * S_ + s] > 0) && valid;
    int nonmark = valid && !mark;
    if (nonmark && !pnm) { span++; nspans++; }
    sid[b * S_ + s] = span < 0 ? 0 : span;
    member[b * S_ + s] = nonmark;
    pnm = nonmark;
  }
  int fb = (nspans == 0);
  if (fb) {
    nspans = 1;
    for (int s = 0; s < S_; ++s) { member[b * S_ + s] = am[b * S_ + s] > 0; sid[b * S_ + s] = 0; }
  }
  nnodes[b] = nspans;
  if (!fb) {
    pnm = 0; int sp = -1;
    for (int s = 0; s < S_; ++s) {
      int valid = am[b * S_ + s] > 0;
      int mark = (spl[b * S_ + s] > 0) && valid;
      int nonmark = valid && !mark;
      if (nonmark && !pnm) sp++;
      if (mark && pnm) {                   // edge group start after a span
        int i = sp, j = sp + 1;
        if (i >= 0 && j < nspans) {        // node_mask (dd) applied here
          if (spl[b * S_ + s] == 1) {
            ga[((size_t)b * NSP + i) * NSP + j] += 1.f;
            deg[b * NSP + i] += 1.f;
          } else if (spl[b * S_ + s] == 2) {
            gp[((size_t)b * NSP + i) * NSP + j] += 1.f;
            gp[((size_t)b * NSP + j) * NSP + i] += 1.f;
            deg[b * NSP + i] += 1.f; deg[b * NSP + j] += 1.f;
          }
        }
      }
      pnm = nonmark;
    }
  }
}

__global__ void seg_sum(const float* __restrict__ seq, const int* __restrict__ sid,
                        const int* __restrict__ member, float* __restrict__ nodes) {
  const int bs = blockIdx.x;
  if (!member[bs]) return;
  const int b = bs / S_;
  int n = sid[bs]; if (n > NSP - 1) n = NSP - 1;
  const float* src = &seq[(size_t)bs * DH];
  float* dst = &nodes[((size_t)b * NPAD + n) * DH];
  for (int d = threadIdx.x; d < DH; d += blockDim.x) atomicAdd(&dst[d], src[d]);
}

__global__ void node_gate(const float* __restrict__ nodes, const float* __restrict__ wv,
                          const float* __restrict__ wb, float* __restrict__ out) {
  __shared__ float red[128];
  int row = blockIdx.x;
  int b = row / NSP, n = row % NSP;
  const float* src = &nodes[((size_t)b * NPAD + n) * DH];
  float s = 0.f;
  for (int d = threadIdx.x; d < DH; d += 128) s += src[d] * wv[d];
  red[threadIdx.x] = s; __syncthreads();
  for (int o = 64; o > 0; o >>= 1) { if (threadIdx.x < o) red[threadIdx.x] += red[threadIdx.x + o]; __syncthreads(); }
  if (!threadIdx.x) out[row] = sigf(red[0] + wb[0]);
}

__global__ __launch_bounds__(256) void gcn_combine(
    const float* __restrict__ selfo, const float* __restrict__ parg,
    const float* __restrict__ ppu, const float* __restrict__ ga,
    const float* __restrict__ gp, const float* __restrict__ wsig,
    const float* __restrict__ deg, float* __restrict__ nodes) {
  __shared__ float sga[NSP], sgp[NSP], sw[NSP];
  int row = blockIdx.x;
  int b = row / NSP, i = row % NSP;
  const float* gar = &ga[((size_t)b * NSP + i) * NSP];
  const float* gpr = &gp[((size_t)b * NSP + i) * NSP];
  for (int j = threadIdx.x; j < NSP; j += 256) {
    sga[j] = gar[j]; sgp[j] = gpr[j]; sw[j] = wsig[b * NSP + j];
  }
  __syncthreads();
  float dg = deg[row]; if (dg < 1.f) dg = 1.f;
  for (int d = threadIdx.x; d < DH; d += 256) {
    float na = 0.f;
    for (int j = 0; j < NSP; ++j) {
      float a = sga[j], p = sgp[j];
      if (a != 0.f || p != 0.f) {
        size_t jr = ((size_t)b * NPAD + j) * DH + d;
        na += sw[j] * (a * parg[jr] + p * ppu[jr]);
      }
    }
    size_t ir = ((size_t)b * NPAD + i) * DH + d;
    float v = selfo[ir] + na / dg;
    nodes[ir] = v > 0.f ? v : 0.f;
  }
}

__global__ __launch_bounds__(256) void fuse_ln(
    const float* __restrict__ seq, const float* __restrict__ nodes,
    const int* __restrict__ sid, const int* __restrict__ member,
    const float* __restrict__ g, const float* __restrict__ bta,
    float* __restrict__ x, u16* __restrict__ xbf) {
  __shared__ float r1[256], r2[256];
  int bs = blockIdx.x;
  int b = bs / S_;
  int n = sid[bs]; if (n > NSP - 1) n = NSP - 1;
  int mem = member[bs];
  const float* sp = &seq[(size_t)bs * DH];
  const float* np = &nodes[((size_t)b * NPAD + n) * DH];
  float v[2], s = 0.f, ss = 0.f;
  for (int k = 0; k < 2; ++k) {
    int d = threadIdx.x + k * 256;
    float val = sp[d] + (mem ? np[d] : 0.f);
    v[k] = val; s += val; ss += val * val;
  }
  r1[threadIdx.x] = s; r2[threadIdx.x] = ss; __syncthreads();
  for (int o = 128; o > 0; o >>= 1) {
    if (threadIdx.x < o) { r1[threadIdx.x] += r1[threadIdx.x + o]; r2[threadIdx.x] += r2[threadIdx.x + o]; }
    __syncthreads();
  }
  float mean = r1[0] / DH;
  float var = r2[0] / DH - mean * mean;
  float inv = rsqrtf(var + 1e-5f);
  for (int k = 0; k < 2; ++k) {
    int d = threadIdx.x + k * 256;
    float o = (v[k] - mean) * inv * g[d] + bta[d];
    x[(size_t)bs * DH + d] = o;
    xbf[(size_t)bs * DH + d] = f2bf(o);
  }
}

__global__ __launch_bounds__(256) void res_ln(
    const float* __restrict__ a, const float* __restrict__ b2,
    const float* __restrict__ g, const float* __restrict__ bta,
    float* __restrict__ y) {
  __shared__ float r1[256], r2[256];
  size_t bs = blockIdx.x;
  float v[2], s = 0.f, ss = 0.f;
  for (int k = 0; k < 2; ++k) {
    int d = threadIdx.x + k * 256;
    float val = a[bs * DH + d] + b2[bs * DH + d];
    v[k] = val; s += val; ss += val * val;
  }
  r1[threadIdx.x] = s; r2[threadIdx.x] = ss; __syncthreads();
  for (int o = 128; o > 0; o >>= 1) {
    if (threadIdx.x < o) { r1[threadIdx.x] += r1[threadIdx.x + o]; r2[threadIdx.x] += r2[threadIdx.x + o]; }
    __syncthreads();
  }
  float mean = r1[0] / DH;
  float var = r2[0] / DH - mean * mean;
  float inv = rsqrtf(var + 1e-5f);
  for (int k = 0; k < 2; ++k) {
    int d = threadIdx.x + k * 256;
    y[bs * DH + d] = (v[k] - mean) * inv * g[d] + bta[d];
  }
}

__global__ void row_dot(const float* __restrict__ x, const float* __restrict__ w,
                        float* __restrict__ out) {
  __shared__ float red[128];
  int bs = blockIdx.x;
  float s = 0.f;
  for (int d = threadIdx.x; d < DH; d += 128) s += x[(size_t)bs * DH + d] * w[d];
  red[threadIdx.x] = s; __syncthreads();
  for (int o = 64; o > 0; o >>= 1) { if (threadIdx.x < o) red[threadIdx.x] += red[threadIdx.x + o]; __syncthreads(); }
  if (!threadIdx.x) out[bs] = red[0];
}

__global__ __launch_bounds__(512) void pool_head(
    const float* __restrict__ scores, const float* __restrict__ mask,
    const float* __restrict__ fused, float* __restrict__ feats, int slot) {
  __shared__ float w[512], red[512];
  int b = blockIdx.x, s = threadIdx.x;
  float m = mask[b * S_ + s];
  float v = scores[b * S_ + s] * m;
  red[s] = v; __syncthreads();
  for (int o = 256; o > 0; o >>= 1) { if (s < o) red[s] = fmaxf(red[s], red[s + o]); __syncthreads(); }
  float mx = red[0]; __syncthreads();
  float e = __expf(v - mx);
  red[s] = e; __syncthreads();
  for (int o = 256; o > 0; o >>= 1) { if (s < o) red[s] += red[s + o]; __syncthreads(); }
  float tot = red[0]; __syncthreads();
  float r = (e / tot) * m;
  red[s] = r; __syncthreads();
  for (int o = 256; o > 0; o >>= 1) { if (s < o) red[s] += red[s + o]; __syncthreads(); }
  float rs = red[0];
  w[s] = r / (rs + 1e-13f);
  __syncthreads();
  float acc = 0.f;
  for (int t = 0; t < S_; ++t) acc += w[t] * fused[((size_t)b * S_ + t) * DH + s];
  feats[b * 1536 + slot * DH + s] = acc;
}

__global__ void copy_cls(const float* __restrict__ fused, float* __restrict__ feats) {
  int b = blockIdx.x;
  for (int d = threadIdx.x; d < DH; d += blockDim.x)
    feats[b * 1536 + 1024 + d] = fused[(size_t)b * S_ * DH + d];
}

__global__ __launch_bounds__(512) void ffn_head(
    const float* __restrict__ feats, const float* __restrict__ w1,
    const float* __restrict__ b1, const float* __restrict__ lng,
    const float* __restrict__ lnb, const float* __restrict__ w2,
    const float* __restrict__ b2, float* __restrict__ out) {
  __shared__ float red[512];
  __shared__ float mv[2];
  int b = blockIdx.x, j = threadIdx.x;
  const float* f = &feats[b * 1536];
  const float* wr = &w1[(size_t)j * 1536];
  float s = b1[j];
  for (int k = 0; k < 1536; ++k) s += f[k] * wr[k];
  red[j] = s; __syncthreads();
  for (int o = 256; o > 0; o >>= 1) { if (j < o) red[j] += red[j + o]; __syncthreads(); }
  if (!j) mv[0] = red[0] / 512.f;
  __syncthreads();
  float d = s - mv[0];
  red[j] = d * d; __syncthreads();
  for (int o = 256; o > 0; o >>= 1) { if (j < o) red[j] += red[j + o]; __syncthreads(); }
  if (!j) mv[1] = red[0] / 512.f;
  __syncthreads();
  float inv = rsqrtf(mv[1] + 1e-5f);
  float x = d * inv * lng[j] + lnb[j];
  float gel = 0.5f * x * (1.f + erff(x * 0.70710678118654752f));
  red[j] = gel * w2[j]; __syncthreads();
  for (int o = 256; o > 0; o >>= 1) { if (j < o) red[j] += red[j + o]; __syncthreads(); }
  if (!j) out[b] = red[0] + b2[0];
}

// ---------------------------------------------------------------------------

extern "C" void kernel_launch(void* const* d_in, const int* in_sizes, int n_in,
                              void* d_out, int out_size, void* d_ws, size_t ws_size,
                              hipStream_t stream) {
  (void)in_sizes; (void)n_in; (void)out_size;
  const int* input_ids = (const int*)d_in[0];
  const int* attn      = (const int*)d_in[1];
  const int* split     = (const int*)d_in[2];
  const float* pmask   = (const float*)d_in[3];
  const float* qmask   = (const float*)d_in[4];
  const float* emb     = (const float*)d_in[5];
  const float* lstm_Wih[2] = { (const float*)d_in[6],  (const float*)d_in[10] };
  const float* lstm_Whh[2] = { (const float*)d_in[7],  (const float*)d_in[11] };
  const float* lstm_bih[2] = { (const float*)d_in[8],  (const float*)d_in[12] };
  const float* lstm_bhh[2] = { (const float*)d_in[9],  (const float*)d_in[13] };
  const float* gcn_w_w = (const float*)d_in[14];
  const float* gcn_w_b = (const float*)d_in[15];
  const float* gcn_self_w = (const float*)d_in[16];
  const float* gcn_self_b = (const float*)d_in[17];
  const float* gcn_arg_w = (const float*)d_in[18];
  const float* gcn_punct_w = (const float*)d_in[19];
  const float* prj_g = (const float*)d_in[20];
  const float* prj_b = (const float*)d_in[21];
  const float* gru_Wih[2][2]; const float* gru_Whh[2][2];
  const float* gru_bih[2][2]; const float* gru_bhh[2][2];
  {
    int idx = 22;
    for (int l = 0; l < 2; ++l)
      for (int d = 0; d < 2; ++d) {
        gru_Wih[l][d] = (const float*)d_in[idx++];
        gru_Whh[l][d] = (const float*)d_in[idx++];
        gru_bih[l][d] = (const float*)d_in[idx++];
        gru_bhh[l][d] = (const float*)d_in[idx++];
      }
  }
  const float* enc_g = (const float*)d_in[38];
  const float* enc_b = (const float*)d_in[39];
  const float* projh = (const float*)d_in[40];
  const float* fc1w  = (const float*)d_in[41];
  const float* fc1b  = (const float*)d_in[42];
  const float* fln_g = (const float*)d_in[43];
  const float* fln_b = (const float*)d_in[44];
  const float* fc2w  = (const float*)d_in[45];
  const float* fc2b  = (const float*)d_in[46];
  float* out = (float*)d_out;

  // ---- workspace layout (bump allocator + one 64MB reused region) ----
  char* base = (char*)d_ws; size_t off = 0;
  auto alloc = [&](size_t bytes) -> char* {
    char* r = base + off; off = (off + bytes + 255) & ~(size_t)255; return r;
  };
  const size_t BS = (size_t)B_ * S_;           // 8192
  u16* emb_bf = (u16*)alloc(BS * E_ * 2);
  u16* wihT[2]; u16* whhT[2]; float* lbias[2];
  for (int d = 0; d < 2; ++d) {
    wihT[d] = (u16*)alloc((size_t)512 * 1024 * 2);
    whhT[d] = (u16*)alloc((size_t)256 * 1024 * 2);
    lbias[d] = (float*)alloc(1024 * 4);
  }
  u16* selfT = (u16*)alloc((size_t)512 * 512 * 2);
  u16* argT  = (u16*)alloc((size_t)512 * 512 * 2);
  u16* punT  = (u16*)alloc((size_t)512 * 512 * 2);
  u16* gwihT[2][2]; u16* gwhhT[2][2];
  for (int l = 0; l < 2; ++l)
    for (int d = 0; d < 2; ++d) {
      gwihT[l][d] = (u16*)alloc((size_t)512 * 768 * 2);
      gwhhT[l][d] = (u16*)alloc((size_t)256 * 768 * 2);
    }
  float* seq  = (float*)alloc(BS * DH * 4);
  float* x    = (float*)alloc(BS * DH * 4);
  u16*   xbf  = (u16*)alloc(BS * DH * 2);
  float* hA   = (float*)alloc(BS * DH * 4);
  float* hB   = (float*)alloc(BS * DH * 4);
  u16*   hbfb = (u16*)alloc(BS * DH * 2);
  float* fused = (float*)alloc(BS * DH * 4);
  int* sid = (int*)alloc(BS * 4);
  int* member = (int*)alloc(BS * 4);
  int* nn = (int*)alloc(64);
  float* ga = (float*)alloc((size_t)B_ * NSP * NSP * 4);
  float* gp = (float*)alloc((size_t)B_ * NSP * NSP * 4);
  float* deg = (float*)alloc((size_t)B_ * NSP * 4);
  float* wsig = (float*)alloc((size_t)B_ * NSP * 4);
  float* nodes = (float*)alloc((size_t)B_ * NPAD * DH * 4);
  u16* nodes_bf = (u16*)alloc((size_t)B_ * NPAD * DH * 2);
  float* scores = (float*)alloc(BS * 4);
  float* feats = (float*)alloc((size_t)B_ * 1536 * 4);
  char* R = alloc((size_t)64 * 1024 * 1024);   // phase-reused region
  if (off > ws_size) return;                   // workspace too small: bail
  float* gi_f = (float*)R;
  float* gi_b = (float*)(R + (size_t)32 * 1024 * 1024);
  float* selfo = (float*)R;
  float* parg  = (float*)(R + (size_t)16 * 1024 * 1024);
  float* ppu   = (float*)(R + (size_t)32 * 1024 * 1024);
  float* ggi_f = (float*)R;
  float* ggi_b = (float*)(R + (size_t)32 * 1024 * 1024);

  auto gemm = [&](const u16* A, const u16* Bm, const float* bias, float* C,
                  u16* Cbf, int M, int Nn, int K) {
    dim3 g(M / 64, Nn / 64);
    gemm_bf16_wmma<<<g, 256, 0, stream>>>(A, Bm, bias, C, Cbf, M, Nn, K);
  };
  auto transp = [&](const float* w, u16* wt, int Nn, int K) {
    size_t n = (size_t)Nn * K;
    transp_bf16<<<(unsigned)((n + 255) / 256), 256, 0, stream>>>(w, wt, Nn, K);
  };

  // ---- 1. weight prep (bf16 transposes, combined biases) ----
  for (int d = 0; d < 2; ++d) {
    transp(lstm_Wih[d], wihT[d], 1024, 512);
    transp(lstm_Whh[d], whhT[d], 1024, 256);
    add_vec<<<4, 256, 0, stream>>>(lstm_bih[d], lstm_bhh[d], lbias[d], 1024);
  }
  transp(gcn_self_w, selfT, 512, 512);
  transp(gcn_arg_w, argT, 512, 512);
  transp(gcn_punct_w, punT, 512, 512);
  for (int l = 0; l < 2; ++l)
    for (int d = 0; d < 2; ++d) {
      transp(gru_Wih[l][d], gwihT[l][d], 768, 512);
      transp(gru_Whh[l][d], gwhhT[l][d], 768, 256);
    }

  // ---- 2. embedding gather -> bf16 ----
  embed_bf<<<(unsigned)BS, 128, 0, stream>>>(input_ids, emb, emb_bf);

  // ---- 3. LSTM input projections (big WMMA GEMMs, biases folded) ----
  gemm(emb_bf, wihT[0], lbias[0], gi_f, nullptr, 8192, 1024, 512);
  gemm(emb_bf, wihT[1], lbias[1], gi_b, nullptr, 8192, 1024, 512);

  // ---- 4. BiLSTM scan (persistent, WMMA h@Whh) ----
  lstm_scan<<<2, 1024, 88 * 1024, stream>>>(gi_f, gi_b, whhT[0], whhT[1], seq);

  // ---- 5. span construction + segment sum ----
  hipMemsetAsync(nodes, 0, (size_t)B_ * NPAD * DH * 4, stream);
  hipMemsetAsync(ga, 0, (size_t)B_ * NSP * NSP * 4, stream);
  hipMemsetAsync(gp, 0, (size_t)B_ * NSP * NSP * 4, stream);
  hipMemsetAsync(deg, 0, (size_t)B_ * NSP * 4, stream);
  span_build<<<B_, 32, 0, stream>>>(attn, split, sid, member, nn, ga, gp, deg);
  seg_sum<<<(unsigned)BS, 256, 0, stream>>>(seq, sid, member, nodes);

  // ---- 6. GCN (2 steps) ----
  const size_t nodeElems = (size_t)B_ * NPAD * DH;
  for (int it = 0; it < 2; ++it) {
    conv_bf16<<<(unsigned)((nodeElems + 255) / 256), 256, 0, stream>>>(nodes, nodes_bf, nodeElems);
    gemm(nodes_bf, selfT, gcn_self_b, selfo, nullptr, B_ * NPAD, 512, 512);
    gemm(nodes_bf, argT, nullptr, parg, nullptr, B_ * NPAD, 512, 512);
    gemm(nodes_bf, punT, nullptr, ppu, nullptr, B_ * NPAD, 512, 512);
    node_gate<<<B_ * NSP, 128, 0, stream>>>(nodes, gcn_w_w, gcn_w_b, wsig);
    gcn_combine<<<B_ * NSP, 256, 0, stream>>>(selfo, parg, ppu, ga, gp, wsig, deg, nodes);
  }

  // ---- 7. gather back + fuse + LN ----
  fuse_ln<<<(unsigned)BS, 256, 0, stream>>>(seq, nodes, sid, member, prj_g, prj_b, x, xbf);

  // ---- 8. 2-layer BiGRU ----
  gemm(xbf, gwihT[0][0], gru_bih[0][0], ggi_f, nullptr, 8192, 768, 512);
  gemm(xbf, gwihT[0][1], gru_bih[0][1], ggi_b, nullptr, 8192, 768, 512);
  gru_scan<<<2, 1024, 72 * 1024, stream>>>(ggi_f, ggi_b, gwhhT[0][0], gwhhT[0][1],
                                           gru_bhh[0][0], gru_bhh[0][1], hA);
  conv_bf16<<<(unsigned)((BS * DH + 255) / 256), 256, 0, stream>>>(hA, hbfb, BS * DH);
  gemm(hbfb, gwihT[1][0], gru_bih[1][0], ggi_f, nullptr, 8192, 768, 512);
  gemm(hbfb, gwihT[1][1], gru_bih[1][1], ggi_b, nullptr, 8192, 768, 512);
  gru_scan<<<2, 1024, 72 * 1024, stream>>>(ggi_f, ggi_b, gwhhT[1][0], gwhhT[1][1],
                                           gru_bhh[1][0], gru_bhh[1][1], hB);

  // ---- 9. residual LN + heads ----
  res_ln<<<(unsigned)BS, 256, 0, stream>>>(hB, x, enc_g, enc_b, fused);
  row_dot<<<(unsigned)BS, 128, 0, stream>>>(fused, projh, scores);
  pool_head<<<B_, 512, 0, stream>>>(scores, pmask, fused, feats, 0);
  pool_head<<<B_, 512, 0, stream>>>(scores, qmask, fused, feats, 1);
  copy_cls<<<B_, 256, 0, stream>>>(fused, feats);
  ffn_head<<<B_, 512, 0, stream>>>(feats, fc1w, fc1b, fln_g, fln_b, fc2w, fc2b, out);
}
